// GraphLearner_73246372266429
// MI455X (gfx1250) — matrix operand order repaired and checked by reading, not compiled, not run on hardware
//
#include <hip/hip_runtime.h>

// ---------------------------------------------------------------------------
// Problem constants (fixed by the reference setup_inputs()).
// ---------------------------------------------------------------------------
#define B_    4
#define T_    8
#define C_    512
#define D_    256
#define KS    4
#define TOPK  10
#define MPD   160.0f
#define NEG_INF (-1.0e20f)

typedef _Float16 half_t;
typedef __attribute__((ext_vector_type(16))) _Float16 v16h;
typedef __attribute__((ext_vector_type(8)))  _Float16 v8h;
typedef __attribute__((ext_vector_type(4)))  _Float16 v4h;
typedef __attribute__((ext_vector_type(2)))  _Float16 v2h;
typedef __attribute__((ext_vector_type(8)))  float    v8f;
typedef __attribute__((ext_vector_type(4)))  float    v4f;

// ---------------------------------------------------------------------------
// Kernel 1: fold wbar into context, produce f16 A (weighted) and B (plain).
// ---------------------------------------------------------------------------
__global__ void convert_kernel(const float* __restrict__ ctx,
                               const float* __restrict__ W,
                               half_t* __restrict__ Ah,
                               half_t* __restrict__ Bh) {
  const int i = (blockIdx.x * blockDim.x + threadIdx.x) * 4;
  const v4f x = *(const v4f*)(ctx + i);
  const int d0 = i & (D_ - 1);

  v4h va, vb;
#pragma unroll
  for (int j = 0; j < 4; ++j) {
    const int d = d0 + j;
    float wb = fmaxf(W[0 * D_ + d], 0.0f)
             + fmaxf(W[1 * D_ + d], 0.0f)
             + fmaxf(W[2 * D_ + d], 0.0f)
             + fmaxf(W[3 * D_ + d], 0.0f);
    wb *= 0.25f;
    va[j] = (half_t)(x[j] * wb);
    vb[j] = (half_t)(x[j]);
  }
  *(v4h*)(Ah + i) = va;
  *(v4h*)(Bh + i) = vb;
}

// ---------------------------------------------------------------------------
// Kernel 2: per (b,t) Gram matrix attn = Ah @ Bh^T.
// Workgroup (256 thr / 8 waves) computes a 64x64 output block. Per 32-wide
// K-step the block stages A(64x32) and B(64x32) f16 panels into LDS with
// global_load_async_to_lds_b128 (one b128 per thread per panel), double-
// buffered through ASYNCcnt so the next panel streams in while WMMAs run.
// Wave w computes a 16x32 sub-block: tile row (w&3), tile cols {2(w>>2),
// 2(w>>2)+1} -> 2 accumulators, 2 v_wmma_f32_16x16x32_f16 per K-step.
// LDS rows padded to 40 halves (80B): 16B-aligned ds_b128 segments and a
// conflict-free bank stride (gcd(20,64)=4 -> 16 distinct banks per group).
// ---------------------------------------------------------------------------
#define BLK       64
#define KSTEP     32
#define STRIDE_H  40                         // padded halves per LDS row
#define PANEL_H   (BLK * STRIDE_H)           // halves per panel (A or B)
#define PANEL_B   (PANEL_H * 2)              // bytes per panel      (5120)
#define BUF_H     (2 * PANEL_H)              // halves per buffer (A+B)

__global__ __launch_bounds__(256)
void gram_wmma_async_kernel(const half_t* __restrict__ Ah,
                            const half_t* __restrict__ Bh,
                            const int* __restrict__ ctx_mask,
                            float* __restrict__ attn) {
  __shared__ half_t lds[2 * BUF_H];          // 20 KiB, double-buffered

  const int bt    = blockIdx.z;              // 0..31
  const int m0    = blockIdx.y * BLK;        // c block
  const int e0    = blockIdx.x * BLK;        // e block
  const int tid   = threadIdx.x;
  const int lane  = tid & 31;
  const int wave  = tid >> 5;
  const int lm    = lane & 15;
  const int group = lane >> 4;
  const int mw    = wave & 3;                // tile row within block
  const int p     = wave >> 2;               // tile-col pair (0 or 1)

  // ---- async staging setup: thread -> (row, 16B segment) ----------------
  const int row = tid >> 2;                  // 0..63
  const int seg = tid & 3;                   // 0..3 (16B segments of 64B row)
  const half_t* gArow = Ah + ((size_t)bt * C_ + (m0 + row)) * D_ + seg * 8;
  const half_t* gBrow = Bh + ((size_t)bt * C_ + (e0 + row)) * D_ + seg * 8;
  // LDS byte offset: flat address truncates to the LDS aperture offset.
  const unsigned ldsBase = (unsigned)(size_t)(void*)lds;
  const unsigned stOff   = ldsBase + (unsigned)(row * (STRIDE_H * 2) + seg * 16);

#define ISSUE_PANELS(bufsel, k0)                                           \
  do {                                                                     \
    unsigned _oA = stOff + (unsigned)((bufsel) * (2 * PANEL_B));           \
    unsigned _oB = _oA + (unsigned)PANEL_B;                                \
    unsigned long long _aA = (unsigned long long)(size_t)(gArow + (k0));   \
    unsigned long long _aB = (unsigned long long)(size_t)(gBrow + (k0));   \
    asm volatile("global_load_async_to_lds_b128 %0, %1, off"               \
                 :: "v"(_oA), "v"(_aA) : "memory");                        \
    asm volatile("global_load_async_to_lds_b128 %0, %1, off"               \
                 :: "v"(_oB), "v"(_aB) : "memory");                        \
  } while (0)

  v8f acc0 = {}, acc1 = {};

  ISSUE_PANELS(0, 0);
  for (int step = 0; step < D_ / KSTEP; ++step) {
    const int buf = step & 1;
    if (step < (D_ / KSTEP) - 1) {
      ISSUE_PANELS(buf ^ 1, (step + 1) * KSTEP);
      asm volatile("s_wait_asynccnt 0x2" ::: "memory");  // this step landed
    } else {
      asm volatile("s_wait_asynccnt 0x0" ::: "memory");
    }
    __syncthreads();

    const half_t* lA = lds + buf * BUF_H;
    const half_t* lB = lA + PANEL_H;

    // A fragment (16x32, row M = mw*16 + lm), CDNA5 16-bit A layout
    const half_t* arow = lA + (mw * 16 + lm) * STRIDE_H;
    v16h a;
#pragma unroll
    for (int r = 0; r < 8; ++r) {
      const int kb = 2 * r + 8 * group + ((r >= 4) ? 8 : 0);
      const v2h pr2 = *(const v2h*)(arow + kb);
      a[2 * r]     = pr2.x;
      a[2 * r + 1] = pr2.y;
    }

    // B fragments (32x16): lane = N, K halves 8*group..+7 and 16+8*group..+7
    const half_t* br0 = lB + ((2 * p)     * 16 + lm) * STRIDE_H + 8 * group;
    const half_t* br1 = lB + ((2 * p + 1) * 16 + lm) * STRIDE_H + 8 * group;
    v16h b0, b1;
    {
      const v8h lo0 = *(const v8h*)br0, hi0 = *(const v8h*)(br0 + 16);
      const v8h lo1 = *(const v8h*)br1, hi1 = *(const v8h*)(br1 + 16);
#pragma unroll
      for (int q = 0; q < 8; ++q) {
        b0[q] = lo0[q]; b0[q + 8] = hi0[q];
        b1[q] = lo1[q]; b1[q + 8] = hi1[q];
      }
    }

    acc0 = __builtin_amdgcn_wmma_f32_16x16x32_f16(false, a, false, b0,
                                                  (short)0, acc0, false, false);
    acc1 = __builtin_amdgcn_wmma_f32_16x16x32_f16(false, a, false, b1,
                                                  (short)0, acc1, false, false);
    __syncthreads();                          // buffer reused two steps later
  }
#undef ISSUE_PANELS

  // ---- mask + store (C/D layout: M = r + 8*group, N = lane%16) ----------
  const int b_idx = bt / T_;
  const int cbase = m0 + mw * 16;
#pragma unroll
  for (int t2 = 0; t2 < 2; ++t2) {
    const int e    = e0 + (2 * p + t2) * 16 + lm;
    const bool e_ok = ctx_mask[b_idx * C_ + e] > 0;
    float* __restrict__ orow = attn + ((size_t)bt * C_ + cbase) * C_ + e;
#pragma unroll
    for (int r = 0; r < 8; ++r) {
      const int  cr   = r + 8 * group;
      const bool c_ok = ctx_mask[b_idx * C_ + cbase + cr] > 0;
      const float v   = (c_ok && e_ok) ? (t2 ? acc1[r] : acc0[r]) : NEG_INF;
      orow[(size_t)cr * C_] = v;
    }
  }
}

// ---------------------------------------------------------------------------
// Kernel 3: one wave per (b,t,c) row: wave-wide top-10, softmax, spatial
// Gaussians, fused zero+scatter in an LDS strip, NT b128 stream-out.
// ---------------------------------------------------------------------------
__global__ __launch_bounds__(32)
void topk_scatter_kernel(const float* __restrict__ attn,
                         const float* __restrict__ mean_dis,
                         const float* __restrict__ prec_dis,
                         float* __restrict__ out) {
  __shared__ v4f tile4[KS * C_ / 4];                 // 8 KiB strip
  float* tile = (float*)tile4;

  const int row  = blockIdx.x;                       // 0 .. B*T*C-1
  const int bt   = row / C_;
  const int c    = row - bt * C_;
  const int lane = threadIdx.x;

  const float* __restrict__ rp = attn + (size_t)row * C_;
  float v[16];
#pragma unroll
  for (int j = 0; j < 16; ++j) v[j] = rp[lane + 32 * j];

  const v4f z4 = {0.0f, 0.0f, 0.0f, 0.0f};
#pragma unroll
  for (int q = 0; q < 16; ++q) tile4[lane + 32 * q] = z4;

  float topv[TOPK];
  int   topi[TOPK];
#pragma unroll
  for (int t = 0; t < TOPK; ++t) {
    float bv = -3.0e38f;
    int   bi = 0x7fffffff;
#pragma unroll
    for (int j = 0; j < 16; ++j) {
      const int idx = lane + 32 * j;
      if (v[j] > bv || (v[j] == bv && idx < bi)) { bv = v[j]; bi = idx; }
    }
#pragma unroll
    for (int off = 16; off > 0; off >>= 1) {
      const float ov = __shfl_xor(bv, off);
      const int   oi = __shfl_xor(bi, off);
      if (ov > bv || (ov == bv && oi < bi)) { bv = ov; bi = oi; }
    }
    topv[t] = bv;
    topi[t] = bi;
    const int owner = bi & 31;
    const int jj    = bi >> 5;
#pragma unroll
    for (int j = 0; j < 16; ++j)
      if (j == jj && lane == owner) v[j] = -3.0e38f;
  }

  const float m = topv[0];
  float sem[TOPK];
  float ssum = 0.0f;
#pragma unroll
  for (int j = 0; j < TOPK; ++j) { sem[j] = __expf(topv[j] - m); ssum += sem[j]; }
  const float sem_inv = 1.0f / ssum;

  if (lane == 0) {
#pragma unroll
    for (int ks = 0; ks < KS; ++ks) {
      const float mu = mean_dis[ks];
      const float pr = prec_dis[ks];
      float sp[TOPK];
      float sps = 0.0f;
#pragma unroll
      for (int j = 0; j < TOPK; ++j) {
        float pc = fminf(fabsf((float)(c - topi[j])), MPD) * (1.0f / MPD);
        const float zz = pc - mu;
        sp[j] = __expf(-0.5f * zz * zz * pr * pr);
        sps += sp[j];
      }
      const float sp_inv = 1.0f / sps;
#pragma unroll
      for (int j = 0; j < TOPK; ++j)
        tile[ks * C_ + topi[j]] = (sem[j] * sem_inv) * (sp[j] * sp_inv);
    }
  }
  __syncthreads();

#pragma unroll
  for (int ks = 0; ks < KS; ++ks) {
    float* __restrict__ orow =
        out + (((size_t)bt * KS + ks) * C_ + c) * (size_t)C_;
#pragma unroll
    for (int q = 0; q < 4; ++q) {
      const int t4 = lane + 32 * q;
      __builtin_nontemporal_store(tile4[ks * (C_ / 4) + t4],
                                  (v4f*)(orow + 4 * t4));
    }
  }
}

// ---------------------------------------------------------------------------
// Host launcher.
//   d_ws layout: [attn 32 MiB][Ah 8 MiB][Bh 8 MiB]
// ---------------------------------------------------------------------------
extern "C" void kernel_launch(void* const* d_in, const int* in_sizes, int n_in,
                              void* d_out, int out_size, void* d_ws, size_t ws_size,
                              hipStream_t stream) {
  const float* ctx  = (const float*)d_in[0];
  const int*   mask = (const int*)d_in[1];
  const float* W    = (const float*)d_in[2];
  const float* mean = (const float*)d_in[3];
  const float* prec = (const float*)d_in[4];
  float*       out  = (float*)d_out;

  char* ws = (char*)d_ws;
  float*  attn = (float*)ws;                                   // 32 MiB
  half_t* Ah   = (half_t*)(ws + (size_t)32 * 1024 * 1024);     //  8 MiB
  half_t* Bh   = (half_t*)(ws + (size_t)40 * 1024 * 1024);     //  8 MiB

  // 1) f32 -> f16 operands with wbar folded in
  {
    const int nThreads = (B_ * T_ * C_ * D_) / 4;              // 1,048,576
    convert_kernel<<<nThreads / 256, 256, 0, stream>>>(ctx, W, Ah, Bh);
  }

  // 2) 32 Gram GEMMs, 64x64 block per workgroup, async double-buffered LDS
  {
    dim3 grid(C_ / BLK, C_ / BLK, B_ * T_);                    // 8 x 8 x 32
    gram_wmma_async_kernel<<<grid, 256, 0, stream>>>(Ah, Bh, mask, attn);
  }

  // 3) top-k + softmax + spatial + fused zero/scatter/stream-out
  {
    topk_scatter_kernel<<<B_ * T_ * C_, 32, 0, stream>>>(attn, mean, prec, out);
  }
}
